// CCTAttention_53781580481156
// MI455X (gfx1250) — matrix operand order repaired and checked by reading, not compiled, not run on hardware
//
#include <hip/hip_runtime.h>
#include <hip/hip_bf16.h>
#include <math.h>
#include <cstdint>

typedef __bf16 bf16;
typedef __bf16 v16bf __attribute__((ext_vector_type(16)));
typedef float  v8f   __attribute__((ext_vector_type(8)));
typedef unsigned int u32x4 __attribute__((ext_vector_type(4)));
typedef int v4i __attribute__((ext_vector_type(4)));

union FragAB { v16bf v; bf16 e[16]; u32x4 q[2]; };
union FragC  { v8f v; float f[8]; };

constexpr int B_ = 2, S_ = 2048, HID_ = 2048;
constexpr int NH_ = 16, NKV_ = 4, HD_ = 128;
constexpr int GROUPS_ = NH_ / NKV_;
constexpr float SCALE_ = 0.08838834764831845f; // HD^-0.5
constexpr int MTOT = B_ * S_; // 4096
constexpr float NEG_BIG = -3.402823466e38f;

// ---------------------------------------------------------------------------
// Async copy helpers (typed v4i pointers, addrspace(1) src / addrspace(3) dst)
// ---------------------------------------------------------------------------
typedef __attribute__((address_space(1))) v4i* gas_v4i_ptr;
typedef __attribute__((address_space(3))) v4i* las_v4i_ptr;

__device__ __forceinline__ void async_copy_b128(const void* g, void* l) {
#if __has_builtin(__builtin_amdgcn_global_load_async_to_lds_b128)
  __builtin_amdgcn_global_load_async_to_lds_b128(
      (gas_v4i_ptr)(uintptr_t)g, (las_v4i_ptr)(uintptr_t)l, 0, 0);
#else
  *(u32x4*)l = *(const u32x4*)g;   // synchronous fallback
#endif
}

__device__ __forceinline__ void wait_async0() {
#if __has_builtin(__builtin_amdgcn_s_wait_asynccnt)
  __builtin_amdgcn_s_wait_asynccnt(0);
#else
  asm volatile("s_wait_asynccnt 0" ::: "memory");
#endif
}

// ---------------------------------------------------------------------------
// f32 -> bf16 convert, 4 elements/thread (b128 loads, b64 stores)
// ---------------------------------------------------------------------------
__global__ void cvt_f32_bf16_v4(const float* __restrict__ in, bf16* __restrict__ out, int n4) {
  int i = blockIdx.x * blockDim.x + threadIdx.x;
  if (i < n4) {
    const float4 v = ((const float4*)in)[i];
    bf16* o = out + 4 * (size_t)i;
    o[0] = (bf16)v.x; o[1] = (bf16)v.y; o[2] = (bf16)v.z; o[3] = (bf16)v.w;
  }
}

// ---------------------------------------------------------------------------
// C[M x N] f32 = A[M x K] bf16 (row major) * W[N x K] bf16 (row major)^T
// block = 128 threads (4 waves), block tile 64x64, wave tile 32x32 (2x2 WMMA)
// A/W 64-wide k-slabs staged in LDS via async-to-LDS, double buffered,
// epilogue peeled so the steady-state loop has no conditional issue.
// ---------------------------------------------------------------------------
__global__ __launch_bounds__(128) void gemm_bf16_nt_async(const bf16* __restrict__ A,
                                                          const bf16* __restrict__ W,
                                                          float* __restrict__ C,
                                                          int M, int N, int K) {
  __shared__ bf16 lA[2][64][64];
  __shared__ bf16 lW[2][64][64];
  const int tid = threadIdx.x;
  const int lane = tid & 31;
  const int wave = tid >> 5;
  const int ln15 = lane & 15;
  const bool hiL = lane >= 16;
  const int m0 = blockIdx.y * 64;
  const int n0 = blockIdx.x * 64;
  const int wm = wave >> 1, wn = wave & 1;

  // Copy mapping: one 64x64 bf16 tile = 64 rows x 128B = 512 x 16B chunks.
  // Thread t copies chunks t, t+128, t+256, t+384: rows r0+16*r, col cc.
  const int r0 = tid >> 3;
  const int cc = (tid & 7) * 8;
  const bf16* gA[4];
  const bf16* gW[4];
#pragma unroll
  for (int r = 0; r < 4; ++r) {
    gA[r] = A + (size_t)(m0 + r0 + 16 * r) * K + cc;
    gW[r] = W + (size_t)(n0 + r0 + 16 * r) * K + cc;
  }

  v8f acc00 = {}, acc01 = {}, acc10 = {}, acc11 = {};
  const int ea = hiL ? 8 : 0;   // A frag K chunks {ea..ea+7, ea+16..ea+23}
  const int eb = hiL ? 16 : 0;  // unused: B frags load the same K chunks as A (ea)

  auto stage = [&](int bufi, int k0) {
#pragma unroll
    for (int r = 0; r < 4; ++r) {
      async_copy_b128(gA[r] + k0, &lA[bufi][r0 + 16 * r][cc]);
      async_copy_b128(gW[r] + k0, &lW[bufi][r0 + 16 * r][cc]);
    }
  };

  auto compute = [&](int bufi) {
#pragma unroll
    for (int ks = 0; ks < 64; ks += 32) {
      FragAB a0, a1, b0, b1;
      const bf16* la0 = &lA[bufi][wm * 32 + ln15][ks];
      const bf16* la1 = la0 + 16 * 64;
      const bf16* lb0 = &lW[bufi][wn * 32 + ln15][ks];
      const bf16* lb1 = lb0 + 16 * 64;
      a0.q[0] = *(const u32x4*)(la0 + ea);
      a0.q[1] = *(const u32x4*)(la0 + ea + 16);
      a1.q[0] = *(const u32x4*)(la1 + ea);
      a1.q[1] = *(const u32x4*)(la1 + ea + 16);
      b0.q[0] = *(const u32x4*)(lb0 + ea);
      b0.q[1] = *(const u32x4*)(lb0 + ea + 16);
      b1.q[0] = *(const u32x4*)(lb1 + ea);
      b1.q[1] = *(const u32x4*)(lb1 + ea + 16);
      acc00 = __builtin_amdgcn_wmma_f32_16x16x32_bf16(false, a0.v, false, b0.v, (short)0, acc00, false, false);
      acc01 = __builtin_amdgcn_wmma_f32_16x16x32_bf16(false, a0.v, false, b1.v, (short)0, acc01, false, false);
      acc10 = __builtin_amdgcn_wmma_f32_16x16x32_bf16(false, a1.v, false, b0.v, (short)0, acc10, false, false);
      acc11 = __builtin_amdgcn_wmma_f32_16x16x32_bf16(false, a1.v, false, b1.v, (short)0, acc11, false, false);
    }
  };

  // prologue: stage k-slab 0 into buffer 0
  stage(0, 0);
  int buf = 0;
  for (int k0 = 0; k0 < K - 64; k0 += 64) {
    wait_async0();
    __syncthreads();
    stage(buf ^ 1, k0 + 64);   // unconditional: epilogue is peeled
    compute(buf);
    buf ^= 1;
  }
  wait_async0();
  __syncthreads();
  compute(buf);                // last slab, nothing left to prefetch

  const int mrow = m0 + wm * 32 + (hiL ? 8 : 0);
  const int ncol = n0 + wn * 32 + ln15;
  FragC c0, c1, c2, c3;
  c0.v = acc00; c1.v = acc01; c2.v = acc10; c3.v = acc11;
#pragma unroll
  for (int e = 0; e < 8; ++e) {
    C[(size_t)(mrow + e) * N + ncol]           = c0.f[e];
    C[(size_t)(mrow + e) * N + ncol + 16]      = c1.f[e];
    C[(size_t)(mrow + 16 + e) * N + ncol]      = c2.f[e];
    C[(size_t)(mrow + 16 + e) * N + ncol + 16] = c3.f[e];
  }
}

// ---------------------------------------------------------------------------
// RoPE on f32 GEMM output (layout [b][s][h][d]) -> bf16 [b][h][s][d]
// ---------------------------------------------------------------------------
__global__ void rope_to_bf16(const float* __restrict__ X, const float* __restrict__ cosp,
                             const float* __restrict__ sinp, bf16* __restrict__ Out,
                             int nheads) {
  const int d = threadIdx.x;           // 0..127
  const int s = blockIdx.x;
  const int h = blockIdx.y;
  const int b = blockIdx.z;
  const size_t src = ((size_t)(b * S_ + s) * nheads + h) * HD_;
  float x = X[src + d];
  float other = X[src + ((d < 64) ? d + 64 : d - 64)];
  float rot = (d < 64) ? -other : other;
  float val = x * cosp[s * HD_ + d] + rot * sinp[s * HD_ + d];
  Out[((size_t)(b * nheads + h) * S_ + s) * HD_ + d] = (bf16)val;
}

// ---------------------------------------------------------------------------
// V f32 [b][s][kv][d] -> bf16 transposed [b][kv][d][s]
// ---------------------------------------------------------------------------
__global__ void v_transpose_bf16(const float* __restrict__ V, bf16* __restrict__ VT) {
  const int d = threadIdx.x;
  const int s = blockIdx.x;
  const int kv = blockIdx.y;
  const int b = blockIdx.z;
  float x = V[((size_t)(b * S_ + s) * NKV_ + kv) * HD_ + d];
  VT[((size_t)(b * NKV_ + kv) * HD_ + d) * S_ + s] = (bf16)x;
}

// ---------------------------------------------------------------------------
// Fused attention per (b, h, 16-row q tile):
//   scores (WMMA) -> LDS, scale*gain + causal mask, softmax, write attn f32,
//   P.V (WMMA) -> ctx f32 [b][s][h*HD+d]
// ---------------------------------------------------------------------------
__global__ __launch_bounds__(128) void attn_fused(
    const bf16* __restrict__ Qb, const bf16* __restrict__ Kb, const bf16* __restrict__ VT,
    const float* __restrict__ pbias, float* __restrict__ attn_out, float* __restrict__ ctx) {
  extern __shared__ float probs[];  // [16][S_]
  const int qt = blockIdx.x, h = blockIdx.y, b = blockIdx.z;
  const int tid = threadIdx.x;
  const int lane = tid & 31, wave = tid >> 5;
  const int ln15 = lane & 15;
  const bool hiL = lane >= 16;
  const int kv = h / GROUPS_;
  const int qs0 = qt * 16;
  const int kcount = (qt + 1) * 16;          // causal: keys <= q
  const int kpad = (kcount + 31) & ~31;      // pad to WMMA K=32
  const int ea = hiL ? 8 : 0;
  const int eb = hiL ? 16 : 0;

  // Q fragments for this tile (HD=128 -> 4 K-slabs of 32)
  const bf16* qrow = Qb + ((size_t)(b * NH_ + h) * S_ + qs0 + ln15) * HD_;
  FragAB aq[4];
#pragma unroll
  for (int kk = 0; kk < 4; ++kk) {
    aq[kk].q[0] = *(const u32x4*)(qrow + kk * 32 + ea);
    aq[kk].q[1] = *(const u32x4*)(qrow + kk * 32 + ea + 16);
  }

  // --- scores: wave w handles key tiles w, w+4, w+8, ... ---
  for (int kt = wave; kt * 16 < kcount; kt += 4) {
    const int kb0 = kt * 16;
    const bf16* krow = Kb + ((size_t)(b * NKV_ + kv) * S_ + kb0 + ln15) * HD_;
    v8f acc = {};
#pragma unroll
    for (int kk = 0; kk < 4; ++kk) {
      FragAB bk;
      bk.q[0] = *(const u32x4*)(krow + kk * 32 + ea);
      bk.q[1] = *(const u32x4*)(krow + kk * 32 + ea + 16);
      acc = __builtin_amdgcn_wmma_f32_16x16x32_bf16(false, aq[kk].v, false, bk.v, (short)0, acc, false, false);
    }
    FragC c; c.v = acc;
    const int key = kb0 + ln15;
#pragma unroll
    for (int e = 0; e < 8; ++e) {
      const int m = e + (hiL ? 8 : 0);
      const int qg = qs0 + m;
      const float gain = 1.0f + pbias[b * S_ + qg];
      float sc = c.f[e] * SCALE_ * gain + ((key <= qg) ? 0.0f : NEG_BIG);
      probs[m * S_ + key] = sc;
    }
  }
  __syncthreads();

  // --- softmax: 8 lanes per row, 16 rows ---
  {
    const int r = tid >> 3, sub = tid & 7;
    const int qg = qs0 + r;
    float mx = NEG_BIG;
    for (int cidx = sub; cidx < kcount; cidx += 8) mx = fmaxf(mx, probs[r * S_ + cidx]);
#pragma unroll
    for (int off = 4; off > 0; off >>= 1) mx = fmaxf(mx, __shfl_xor(mx, off, 8));
    float sum = 0.0f;
    for (int cidx = sub; cidx < kcount; cidx += 8) {
      float ev = __expf(probs[r * S_ + cidx] - mx);
      probs[r * S_ + cidx] = ev;
      sum += ev;
    }
#pragma unroll
    for (int off = 4; off > 0; off >>= 1) sum += __shfl_xor(sum, off, 8);
    const float inv = 1.0f / sum;
    float* arow = attn_out + ((size_t)(b * NH_ + h) * S_ + qg) * S_;
    for (int cidx = sub; cidx < kcount; cidx += 8) {
      float p = probs[r * S_ + cidx] * inv;
      probs[r * S_ + cidx] = p;
      arow[cidx] = p;
    }
    for (int cidx = kcount + sub; cidx < kpad; cidx += 8) probs[r * S_ + cidx] = 0.0f;
    for (int cidx = kcount + sub; cidx < S_; cidx += 8) arow[cidx] = 0.0f;  // masked-out keys
  }
  __syncthreads();

  // --- P.V: wave w owns output dims [w*32, w*32+32) ---
  {
    const int d0 = wave * 32;
    v8f c0 = {}, c1 = {};
    const bf16* vt0 = VT + ((size_t)(b * NKV_ + kv) * HD_ + d0 + ln15) * S_;
    const bf16* vt1 = vt0 + (size_t)16 * S_;
    for (int kb = 0; kb < kpad; kb += 32) {
      FragAB ap;
      const float* pr = &probs[ln15 * S_ + kb + (hiL ? 8 : 0)];
#pragma unroll
      for (int j = 0; j < 8; ++j) {
        ap.e[j]     = (bf16)pr[j];
        ap.e[8 + j] = (bf16)pr[16 + j];
      }
      FragAB bv0, bv1;
      bv0.q[0] = *(const u32x4*)(vt0 + kb + ea);
      bv0.q[1] = *(const u32x4*)(vt0 + kb + ea + 16);
      bv1.q[0] = *(const u32x4*)(vt1 + kb + ea);
      bv1.q[1] = *(const u32x4*)(vt1 + kb + ea + 16);
      c0 = __builtin_amdgcn_wmma_f32_16x16x32_bf16(false, ap.v, false, bv0.v, (short)0, c0, false, false);
      c1 = __builtin_amdgcn_wmma_f32_16x16x32_bf16(false, ap.v, false, bv1.v, (short)0, c1, false, false);
    }
    FragC f0, f1; f0.v = c0; f1.v = c1;
#pragma unroll
    for (int e = 0; e < 8; ++e) {
      const int qg = qs0 + e + (hiL ? 8 : 0);
      float* crow = ctx + ((size_t)b * S_ + qg) * HID_ + h * HD_;
      crow[d0 + ln15]      = f0.f[e];
      crow[d0 + 16 + ln15] = f1.f[e];
    }
  }
}

// ---------------------------------------------------------------------------
extern "C" void kernel_launch(void* const* d_in, const int* in_sizes, int n_in,
                              void* d_out, int out_size, void* d_ws, size_t ws_size,
                              hipStream_t stream) {
  (void)in_sizes; (void)n_in; (void)out_size; (void)ws_size;
  const float* hs    = (const float*)d_in[0];
  const float* wq    = (const float*)d_in[1];
  const float* wk    = (const float*)d_in[2];
  const float* wv    = (const float*)d_in[3];
  const float* wo    = (const float*)d_in[4];
  const float* cosp  = (const float*)d_in[5];
  const float* sinp  = (const float*)d_in[6];
  /* d_in[7]: attention_mask (causal; reconstructed analytically) */
  const float* pbias = (const float*)d_in[8];

  char* ws = (char*)d_ws;
  size_t off = 0;
  auto alloc = [&](size_t bytes) -> char* {
    char* p = ws + off;
    off += (bytes + 255) & ~(size_t)255;
    return p;
  };
  bf16*  hsb  = (bf16*)alloc((size_t)MTOT * HID_ * 2);
  bf16*  wqb  = (bf16*)alloc((size_t)HID_ * HID_ * 2);
  bf16*  wkb  = (bf16*)alloc((size_t)NKV_ * HD_ * HID_ * 2);
  bf16*  wvb  = (bf16*)alloc((size_t)NKV_ * HD_ * HID_ * 2);
  bf16*  wob  = (bf16*)alloc((size_t)HID_ * HID_ * 2);
  float* qf   = (float*)alloc((size_t)MTOT * NH_ * HD_ * 4);
  float* kf   = (float*)alloc((size_t)MTOT * NKV_ * HD_ * 4);
  float* vf   = (float*)alloc((size_t)MTOT * NKV_ * HD_ * 4);
  bf16*  qb   = (bf16*)alloc((size_t)MTOT * NH_ * HD_ * 2);
  bf16*  kbuf = (bf16*)alloc((size_t)MTOT * NKV_ * HD_ * 2);
  bf16*  vtb  = (bf16*)alloc((size_t)MTOT * NKV_ * HD_ * 2);
  float* ctx  = (float*)alloc((size_t)MTOT * HID_ * 4);
  bf16*  ctxb = (bf16*)alloc((size_t)MTOT * HID_ * 2);

  float* out_o    = (float*)d_out;                    // (B, S, HID)
  float* out_attn = out_o + (size_t)B_ * S_ * HID_;   // (B, NH, S, S)

  auto cvt = [&](const float* src, bf16* dst, int n) {
    cvt_f32_bf16_v4<<<(n / 4 + 255) / 256, 256, 0, stream>>>(src, dst, n / 4);
  };
  cvt(hs, hsb, MTOT * HID_);
  cvt(wq, wqb, HID_ * HID_);
  cvt(wk, wkb, NKV_ * HD_ * HID_);
  cvt(wv, wvb, NKV_ * HD_ * HID_);
  cvt(wo, wob, HID_ * HID_);

  // projections: C = A . W^T (LDS-staged async double-buffered WMMA GEMM)
  gemm_bf16_nt_async<<<dim3(HID_ / 64, MTOT / 64), 128, 0, stream>>>(hsb, wqb, qf, MTOT, HID_, HID_);
  gemm_bf16_nt_async<<<dim3((NKV_ * HD_) / 64, MTOT / 64), 128, 0, stream>>>(hsb, wkb, kf, MTOT, NKV_ * HD_, HID_);
  gemm_bf16_nt_async<<<dim3((NKV_ * HD_) / 64, MTOT / 64), 128, 0, stream>>>(hsb, wvb, vf, MTOT, NKV_ * HD_, HID_);

  rope_to_bf16<<<dim3(S_, NH_, B_), HD_, 0, stream>>>(qf, cosp, sinp, qb, NH_);
  rope_to_bf16<<<dim3(S_, NKV_, B_), HD_, 0, stream>>>(kf, cosp, sinp, kbuf, NKV_);
  v_transpose_bf16<<<dim3(S_, NKV_, B_), HD_, 0, stream>>>(vf, vtb);

  attn_fused<<<dim3(S_ / 16, NH_, B_), 128, 16 * S_ * sizeof(float), stream>>>(
      qb, kbuf, vtb, pbias, out_attn, ctx);

  cvt(ctx, ctxb, MTOT * HID_);
  gemm_bf16_nt_async<<<dim3(HID_ / 64, MTOT / 64), 128, 0, stream>>>(ctxb, wob, out_o, MTOT, HID_, HID_);
}
